// Attention_7447473291775
// MI455X (gfx1250) — matrix operand order repaired and checked by reading, not compiled, not run on hardware
//
#include <hip/hip_runtime.h>

// ---------------------------------------------------------------------------
// Attention (Q = h@W_b, K = V = b), flash-style, CDNA5 bf16 WMMA.
//   B=8, S_enc=S_dec=4096, D=128.
// ~1400 flop/byte -> compute bound; all matmul on v_wmma_f32_16x16x32_bf16.
// 64-key blocks; DPP16 VALU reductions; log2-domain softmax; wave-uniform
// skip of the O-rescale when no row max increases.
// ---------------------------------------------------------------------------

typedef __bf16 bf16;
typedef __attribute__((ext_vector_type(16))) __bf16 v16bf;
typedef __attribute__((ext_vector_type(8)))  __bf16 v8bf;
typedef __attribute__((ext_vector_type(8)))  float  v8f;

#define NB     8
#define S_ENC  4096
#define S_DEC  4096
#define DDIM   128
#define LOG2E  1.4426950408889634f

union V16 { v16bf v; v8bf h[2]; };

// LDS row strides (elements): multiples of 8 elems (16B) keep ds_load_b128
// alignment; pads break the 256B bank period.
#define KSTRIDE 72    // K^T tile: 128 d-rows x 64 keys   (18432 B)
#define VSTRIDE 136   // V tile:   64 key-rows x 128 d    (17408 B)
#define PSTRIDE 72    // per-wave P tile: 16 x 64         (18432 B)

// ---- 16-lane butterfly reductions in the VALU (DPP16) --------------------
// xor1: quad_perm[1,0,3,2]=0xB1; xor2: quad_perm[2,3,0,1]=0x4E;
// xor7: ROW_HALF_MIRROR=0x141;   xor15: ROW_MIRROR=0x140.
__device__ __forceinline__ float red_max16(float v) {
  int t;
  t = __builtin_amdgcn_update_dpp(0, __float_as_int(v), 0xB1, 0xf, 0xf, true);
  v = fmaxf(v, __int_as_float(t));
  t = __builtin_amdgcn_update_dpp(0, __float_as_int(v), 0x4E, 0xf, 0xf, true);
  v = fmaxf(v, __int_as_float(t));
  t = __builtin_amdgcn_update_dpp(0, __float_as_int(v), 0x141, 0xf, 0xf, true);
  v = fmaxf(v, __int_as_float(t));
  t = __builtin_amdgcn_update_dpp(0, __float_as_int(v), 0x140, 0xf, 0xf, true);
  v = fmaxf(v, __int_as_float(t));
  return v;
}
__device__ __forceinline__ float red_add16(float v) {
  int t;
  t = __builtin_amdgcn_update_dpp(0, __float_as_int(v), 0xB1, 0xf, 0xf, true);
  v += __int_as_float(t);
  t = __builtin_amdgcn_update_dpp(0, __float_as_int(v), 0x4E, 0xf, 0xf, true);
  v += __int_as_float(t);
  t = __builtin_amdgcn_update_dpp(0, __float_as_int(v), 0x141, 0xf, 0xf, true);
  v += __int_as_float(t);
  t = __builtin_amdgcn_update_dpp(0, __float_as_int(v), 0x140, 0xf, 0xf, true);
  v += __int_as_float(t);
  return v;
}

// ---------------------------------------------------------------------------
// Prep 1: Q = h @ W_b (f32 accumulate, bf16 out). ~1 GFLOP, negligible.
// ---------------------------------------------------------------------------
__global__ void qproj_kernel(const float* __restrict__ hsrc,
                             const float* __restrict__ W,
                             bf16* __restrict__ Q) {
  __shared__ float hrow[DDIM];
  const int row = blockIdx.x;
  const int c   = threadIdx.x;          // 0..127
  hrow[c] = hsrc[(size_t)row * DDIM + c];
  __syncthreads();
  float acc = 0.f;
#pragma unroll 8
  for (int d = 0; d < DDIM; ++d) acc = fmaf(hrow[d], W[d * DDIM + c], acc);
  Q[(size_t)row * DDIM + c] = (bf16)acc;
}

// ---------------------------------------------------------------------------
// Prep 2: b -> V (bf16 row-major) and K^T (bf16 [d][s]) via 32x32 LDS tiles.
// ---------------------------------------------------------------------------
__global__ void kv_prep_kernel(const float* __restrict__ bsrc,
                               bf16* __restrict__ KT,
                               bf16* __restrict__ V) {
  __shared__ bf16 tile[32][33];
  const int batch = blockIdx.z;
  const int s0 = blockIdx.x * 32;
  const int d0 = blockIdx.y * 32;
  const int tx = threadIdx.x;           // 0..31
  const int ty = threadIdx.y;           // 0..7
#pragma unroll
  for (int i = 0; i < 4; ++i) {
    const int s = s0 + ty + i * 8, d = d0 + tx;
    const float x = bsrc[((size_t)batch * S_ENC + s) * DDIM + d];
    const bf16 bx = (bf16)x;
    tile[ty + i * 8][tx] = bx;
    V[((size_t)batch * S_ENC + s) * DDIM + d] = bx;
  }
  __syncthreads();
#pragma unroll
  for (int i = 0; i < 4; ++i) {
    const int d = d0 + ty + i * 8, s = s0 + tx;
    KT[((size_t)batch * DDIM + d) * S_ENC + s] = tile[tx][ty + i * 8];
  }
}

// ---------------------------------------------------------------------------
// Flash attention. 256 threads = 8 waves; each wave owns 16 query rows.
// Per 64-key block: cooperative K^T/V LDS tiles, 16 score WMMAs, f32 online
// softmax (log2 domain, DPP reductions, lazy rescale), P C->A transpose
// through per-wave LDS, 16 PV WMMAs. 64 iterations over S_enc.
// ---------------------------------------------------------------------------
__global__ __launch_bounds__(256)
void flash_attn_kernel(const bf16* __restrict__ Q,
                       const bf16* __restrict__ KT,
                       const bf16* __restrict__ V,
                       float* __restrict__ out) {
  __shared__ bf16 kt[128 * KSTRIDE];        // [d][key]
  __shared__ bf16 vt[64 * VSTRIDE];         // [key][d]
  __shared__ bf16 pt[8 * 16 * PSTRIDE];     // per-wave P

  const int batch = blockIdx.y;
  const int qtile = blockIdx.x;
  const int tid  = threadIdx.x;
  const int wave = tid >> 5;
  const int lane = tid & 31;
  const int half = lane >> 4;
  const int ln   = lane & 15;

  const int qrow0 = qtile * 128 + wave * 16;

  // Q A-fragments (16x32 bf16, 4 chunks over d)
  V16 qa[4];
  {
    const bf16* qrow = Q + ((size_t)batch * S_DEC + qrow0 + ln) * DDIM;
#pragma unroll
    for (int c = 0; c < 4; ++c) {
      qa[c].h[0] = *(const v8bf*)(qrow + c * 32 + half * 8);
      qa[c].h[1] = *(const v8bf*)(qrow + c * 32 + 16 + half * 8);
    }
  }

  const v8f vzero = {0.f, 0.f, 0.f, 0.f, 0.f, 0.f, 0.f, 0.f};
  v8f o[8];
#pragma unroll
  for (int j = 0; j < 8; ++j) o[j] = vzero;

  // m2row = running row-max scaled by log2(e)  (softmax done in exp2 domain)
  float m2row[8], lrow[8];
#pragma unroll
  for (int r = 0; r < 8; ++r) { m2row[r] = -3.0e38f; lrow[r] = 0.f; }

  const bf16* KTb = KT + (size_t)batch * DDIM * S_ENC;
  const bf16* Vb  = V  + (size_t)batch * S_ENC * DDIM;
  bf16* ptw = pt + wave * 16 * PSTRIDE;

  for (int kb = 0; kb < S_ENC; kb += 64) {
    __syncthreads();                         // tiles free for overwrite
    {  // K^T tile: 128 rows x 64 keys; 2 threads/row, 64B each
      const int d = tid >> 1, hh = (tid & 1) * 32;
      const bf16* src = KTb + (size_t)d * S_ENC + kb + hh;
      bf16* dst = kt + d * KSTRIDE + hh;
#pragma unroll
      for (int u = 0; u < 4; ++u)
        *(v8bf*)(dst + u * 8) = *(const v8bf*)(src + u * 8);
      // V tile: 64 rows x 128 d; 4 threads/row, 64B each
      const int kk = tid >> 2, p = (tid & 3) * 32;
      const bf16* vsrc = Vb + (size_t)(kb + kk) * DDIM + p;
      bf16* vdst = vt + kk * VSTRIDE + p;
#pragma unroll
      for (int u = 0; u < 4; ++u)
        *(v8bf*)(vdst + u * 8) = *(const v8bf*)(vsrc + u * 8);
    }
    __syncthreads();

    // ---- scores: S[16x64] = Q(16x128) * K^T(128x64): 4 C tiles, 16 WMMAs
    v8f s[4];
#pragma unroll
    for (int t = 0; t < 4; ++t) s[t] = vzero;
#pragma unroll
    for (int c = 0; c < 4; ++c) {
      const bf16* krow = kt + (c * 32 + half * 8) * KSTRIDE;  // lane = key column, elems over d
#pragma unroll
      for (int t = 0; t < 4; ++t) {
        V16 bt;
#pragma unroll
        for (int e = 0; e < 8; ++e) {
          bt.v[e]     = krow[e * KSTRIDE + t * 16 + ln];
          bt.v[8 + e] = krow[(16 + e) * KSTRIDE + t * 16 + ln];
        }
        s[t] = __builtin_amdgcn_wmma_f32_16x16x32_bf16(
            false, qa[c].v, false, bt.v, (short)0, s[t], false, false);
      }
    }

    // ---- online softmax, exp2 domain; row M=r+8*half lives across 16 lanes
    float pm[8];
#pragma unroll
    for (int r = 0; r < 8; ++r)
      pm[r] = fmaxf(fmaxf(s[0][r], s[1][r]), fmaxf(s[2][r], s[3][r]));
#pragma unroll
    for (int r = 0; r < 8; ++r) pm[r] = red_max16(pm[r]) * LOG2E;

    int raise = 0;
#pragma unroll
    for (int r = 0; r < 8; ++r) raise |= (pm[r] > m2row[r]) ? 1 : 0;
    if (__any(raise)) {         // wave-uniform: rescale only when a max moved
      float alpha[8];
#pragma unroll
      for (int r = 0; r < 8; ++r) {
        const float mn = fmaxf(m2row[r], pm[r]);
        alpha[r] = __builtin_amdgcn_exp2f(m2row[r] - mn);
        m2row[r] = mn;
        lrow[r] *= alpha[r];
      }
#pragma unroll
      for (int j = 0; j < 8; ++j)
#pragma unroll
        for (int r = 0; r < 8; ++r) o[j][r] *= alpha[r];
    }

    // P = exp2(s*log2e - m2) : one fma + one exp per element
    float negm[8], ps[8];
#pragma unroll
    for (int r = 0; r < 8; ++r) negm[r] = -m2row[r];
#pragma unroll
    for (int t = 0; t < 4; ++t)
#pragma unroll
      for (int r = 0; r < 8; ++r)
        s[t][r] = __builtin_amdgcn_exp2f(fmaf(s[t][r], LOG2E, negm[r]));
#pragma unroll
    for (int r = 0; r < 8; ++r)
      ps[r] = (s[0][r] + s[1][r]) + (s[2][r] + s[3][r]);
#pragma unroll
    for (int r = 0; r < 8; ++r) lrow[r] += red_add16(ps[r]);

    // ---- P: C layout -> A layout via per-wave LDS tile
#pragma unroll
    for (int r = 0; r < 8; ++r) {
      bf16* base = ptw + (r + half * 8) * PSTRIDE + ln;
      base[0]  = (bf16)s[0][r];
      base[16] = (bf16)s[1][r];
      base[32] = (bf16)s[2][r];
      base[48] = (bf16)s[3][r];
    }
    asm volatile("s_wait_dscnt 0" ::: "memory");   // same-wave DS ordering
    V16 pa0, pa1;
    {
      const bf16* prow = ptw + ln * PSTRIDE;
      pa0.h[0] = *(const v8bf*)(prow + half * 8);
      pa0.h[1] = *(const v8bf*)(prow + 16 + half * 8);
      pa1.h[0] = *(const v8bf*)(prow + 32 + half * 8);
      pa1.h[1] = *(const v8bf*)(prow + 48 + half * 8);
    }

    // ---- O[16x128] += P(16x64) * V(64x128): 16 WMMAs
#pragma unroll
    for (int j = 0; j < 8; ++j) {
      V16 bv0, bv1;
      const bf16* v0 = vt + (half * 8) * VSTRIDE + j * 16 + ln;         // keys 0..31
      const bf16* v1 = vt + (32 + half * 8) * VSTRIDE + j * 16 + ln;  // keys 32..63
#pragma unroll
      for (int e = 0; e < 8; ++e) {
        bv0.v[e]     = v0[e * VSTRIDE];
        bv0.v[8 + e] = v0[(16 + e) * VSTRIDE];
        bv1.v[e]     = v1[e * VSTRIDE];
        bv1.v[8 + e] = v1[(16 + e) * VSTRIDE];
      }
      o[j] = __builtin_amdgcn_wmma_f32_16x16x32_bf16(
          false, pa0.v, false, bv0.v, (short)0, o[j], false, false);
      o[j] = __builtin_amdgcn_wmma_f32_16x16x32_bf16(
          false, pa1.v, false, bv1.v, (short)0, o[j], false, false);
    }
  }

  // ---- normalize and store f32
  float* obase = out + ((size_t)batch * S_DEC + qrow0) * DDIM;
#pragma unroll
  for (int r = 0; r < 8; ++r) {
    const float inv = 1.0f / lrow[r];
    const int M = r + half * 8;
#pragma unroll
    for (int j = 0; j < 8; ++j)
      obase[(size_t)M * DDIM + j * 16 + ln] = o[j][r] * inv;
  }
}

// ---------------------------------------------------------------------------
extern "C" void kernel_launch(void* const* d_in, const int* in_sizes, int n_in,
                              void* d_out, int out_size, void* d_ws, size_t ws_size,
                              hipStream_t stream) {
  const float* bmat = (const float*)d_in[0];   // [B,S_enc,D] f32
  const float* hmat = (const float*)d_in[1];   // [B,S_dec,D] f32
  const float* Wb   = (const float*)d_in[2];   // [D,D] f32
  float* out = (float*)d_out;                  // [B,S_dec,D] f32

  bf16* Q  = (bf16*)d_ws;                         // 8 MB
  bf16* KT = Q  + (size_t)NB * S_DEC * DDIM;      // 8 MB
  bf16* V  = KT + (size_t)NB * DDIM * S_ENC;      // 8 MB

  qproj_kernel<<<NB * S_DEC, 128, 0, stream>>>(hmat, Wb, Q);

  dim3 g2(S_ENC / 32, DDIM / 32, NB), b2(32, 8);
  kv_prep_kernel<<<g2, b2, 0, stream>>>(bmat, KT, V);

  dim3 g3(S_DEC / 128, NB), b3(256);
  flash_attn_kernel<<<g3, b3, 0, stream>>>(Q, KT, V, out);
}
